// PairClassifier_82592221102740
// MI455X (gfx1250) — matrix operand order repaired and checked by reading, not compiled, not run on hardware
//
#include <hip/hip_runtime.h>

typedef __attribute__((ext_vector_type(16))) _Float16 v16h;
typedef __attribute__((ext_vector_type(8)))  _Float16 v8h;
typedef __attribute__((ext_vector_type(8)))  float    v8f;
typedef __attribute__((ext_vector_type(4)))  float    v4f;

#define WAVES    4
#define TILES    2     // 16-pair M tiles per wave (B-fragment reuse factor)
#define RS_FEAT  520   // 512 + 8 halves pad -> 4-bank row skew
#define RS_W1    520
#define RS_W2    136   // 128 + 8 pad
#define RS_X1    136

// Prepped weight image (f16, transposed, padded rows) in device-global memory
#define WS_W1T   0
#define WS_W2T   (128 * RS_W1)            // 66560
#define WS_TOTAL (WS_W2T + 64 * RS_W2)    // 75264 halves = 147 KB

// LDS layout, in halves
#define OFF_W1T  0
#define OFF_W2T  (OFF_W1T + 128 * RS_W1)                    // 66560
#define OFF_FEAT (OFF_W2T + 64 * RS_W2)                     // 75264
#define OFF_X1   (OFF_FEAT + WAVES * TILES * 16 * RS_FEAT)  // 141824
#define OFF_END  (OFF_X1 + WAVES * TILES * 16 * RS_X1)      // 159232 halves
#define LDS_BYTES ((size_t)OFF_END * 2 + 320 * 4)           // 319744 B <= 320 KB

__device__ __align__(16) _Float16 g_wsw[WS_TOTAL];

static __device__ __forceinline__ v16h comb16(v8h lo, v8h hi) {
    return __builtin_shufflevector(lo, hi, 0,1,2,3,4,5,6,7,8,9,10,11,12,13,14,15);
}

// ---- Prep: W1 (512x128 f32) -> w1t f16 [128][520]; W2 (128x64) -> w2t f16 [64][136]
extern "C" __global__ void prep_weights(const float* __restrict__ W1,
                                        const float* __restrict__ W2) {
    int idx = blockIdx.x * blockDim.x + threadIdx.x;
    const int stride = gridDim.x * blockDim.x;
    for (; idx < WS_TOTAL; idx += stride) {
        _Float16 v;
        if (idx < WS_W2T) {
            const int n = idx / RS_W1, k = idx % RS_W1;
            v = (k < 512) ? (_Float16)W1[(size_t)k * 128 + n] : (_Float16)0.0f;
        } else {
            const int j = idx - WS_W2T;
            const int n = j / RS_W2, k = j % RS_W2;
            v = (k < 128) ? (_Float16)W2[k * 64 + n] : (_Float16)0.0f;
        }
        g_wsw[idx] = v;
    }
}

extern "C" __global__ void __launch_bounds__(WAVES * 32)
pair_mlp_wmma(const float* __restrict__ H,
              const long long* __restrict__ PAIRS,
              const float* __restrict__ B1,
              const float* __restrict__ B2,
              const float* __restrict__ W3, const float* __restrict__ B3,
              float* __restrict__ OUT, int n_pairs)
{
    extern __shared__ _Float16 lds[];
    _Float16* w1t  = lds + OFF_W1T;   // [128][RS_W1]
    _Float16* w2t  = lds + OFF_W2T;   // [64][RS_W2]
    _Float16* feat = lds + OFF_FEAT;  // [WAVES*TILES][16][RS_FEAT]
    _Float16* x1s  = lds + OFF_X1;    // [WAVES*TILES][16][RS_X1]
    float* fbuf = (float*)(lds + OFF_END);
    float* sb1 = fbuf;          // 128
    float* sb2 = fbuf + 128;    // 64
    float* sw3 = fbuf + 192;    // 64
    float* sb3 = fbuf + 256;    // 1

    const int tid = threadIdx.x;

    // ---- Stage prepped f16 weights: straight coalesced b128 copy ----
    {
        #pragma unroll 4
        for (int c8 = tid; c8 < WS_TOTAL / 8; c8 += WAVES * 32)
            *(v8h*)(lds + (size_t)c8 * 8) = *(const v8h*)(g_wsw + (size_t)c8 * 8);
        if (tid < 64) { sb2[tid] = B2[tid]; sw3[tid] = W3[tid]; }
        sb1[tid] = B1[tid];
        if (tid == 0) sb3[0] = B3[0];
    }
    __syncthreads();

    const int wave = tid >> 5;
    const int lane = tid & 31;
    const int sub  = lane & 15;   // row M within tile / column N within n-tile
    const int hfid = lane >> 4;   // half-wave id
    const int wtile = (blockIdx.x * WAVES + wave) * TILES;

    // ---- Gather pair rows, build feature tiles [hs | ht | hs*ht | |hs-ht|] ----
    #pragma unroll
    for (int tt = 0; tt < TILES; ++tt) {
        const int tbase = (wtile + tt) * 16;
        const int p  = tbase + sub;
        const int pc = p < n_pairs ? p : (n_pairs - 1);
        const long long si = PAIRS[2 * (size_t)pc];
        const long long ti = PAIRS[2 * (size_t)pc + 1];
        const float* hs = H + (size_t)si * 128;
        const float* ht = H + (size_t)ti * 128;
        _Float16* frow = feat + ((wave * TILES + tt) * 16 + sub) * RS_FEAT;
        const int j0 = hfid * 64;
        #pragma unroll
        for (int j = 0; j < 64; j += 8) {
            v4f a0 = *(const v4f*)(hs + j0 + j);
            v4f a1 = *(const v4f*)(hs + j0 + j + 4);
            v4f b0 = *(const v4f*)(ht + j0 + j);
            v4f b1 = *(const v4f*)(ht + j0 + j + 4);
            v8h fs, ft, fp, fd;
            #pragma unroll
            for (int e = 0; e < 4; ++e) {
                float xa = a0[e], xb = b0[e];
                fs[e] = (_Float16)xa;
                ft[e] = (_Float16)xb;
                fp[e] = (_Float16)(xa * xb);
                fd[e] = (_Float16)fabsf(xa - xb);
                float ya = a1[e], yb = b1[e];
                fs[e + 4] = (_Float16)ya;
                ft[e + 4] = (_Float16)yb;
                fp[e + 4] = (_Float16)(ya * yb);
                fd[e + 4] = (_Float16)fabsf(ya - yb);
            }
            *(v8h*)(frow +       j0 + j) = fs;
            *(v8h*)(frow + 128 + j0 + j) = ft;
            *(v8h*)(frow + 256 + j0 + j) = fp;
            *(v8h*)(frow + 384 + j0 + j) = fd;
        }
    }
    __syncthreads();

    const _Float16* ar1[TILES];
    #pragma unroll
    for (int tt = 0; tt < TILES; ++tt)
        ar1[tt] = feat + ((wave * TILES + tt) * 16 + sub) * RS_FEAT;
    _Float16* xw = x1s + (wave * TILES) * 16 * RS_X1;

    // ---- Layer 1: [16x512]@[512x128], 2 M-tiles share each B fragment.
    //      Two N-passes of 4 n-tiles keep accumulators at 2x4x8 = 64 VGPRs. ----
    #pragma unroll
    for (int pass = 0; pass < 2; ++pass) {
        v8f acc[TILES][4];
        #pragma unroll
        for (int tt = 0; tt < TILES; ++tt)
            #pragma unroll
            for (int nt = 0; nt < 4; ++nt) acc[tt][nt] = (v8f){};

        const _Float16* br[4];
        #pragma unroll
        for (int nt = 0; nt < 4; ++nt)
            br[nt] = w1t + ((pass * 4 + nt) * 16 + sub) * RS_W1;

        #pragma unroll
        for (int kk = 0; kk < 16; ++kk) {
            // A layout (ISA 16-bit A 16x32): hw0 -> K {0..7,16..23}, hw1 -> {8..15,24..31}
            const int ka = kk * 32 + hfid * 8;
            v16h a0 = comb16(*(const v8h*)(ar1[0] + ka),
                             *(const v8h*)(ar1[0] + ka + 16));
            v16h a1 = comb16(*(const v8h*)(ar1[1] + ka),
                             *(const v8h*)(ar1[1] + ka + 16));
            // B layout (matches A): hw0 -> K {0..7,16..23}, hw1 -> {8..15,24..31}
            const int kb = kk * 32 + hfid * 8;
            #pragma unroll
            for (int nt = 0; nt < 4; ++nt) {
                v16h b = comb16(*(const v8h*)(br[nt] + kb),
                                *(const v8h*)(br[nt] + kb + 16));
                acc[0][nt] = __builtin_amdgcn_wmma_f32_16x16x32_f16(
                    false, a0, false, b, (short)0, acc[0][nt], false, false);
                acc[1][nt] = __builtin_amdgcn_wmma_f32_16x16x32_f16(
                    false, a1, false, b, (short)0, acc[1][nt], false, false);
            }
        }

        // bias + ReLU -> f16 x1 tiles (D: VGPR r -> M=r+8*hfid, lane -> N)
        #pragma unroll
        for (int nt = 0; nt < 4; ++nt) {
            const int n = (pass * 4 + nt) * 16 + sub;
            const float bias = sb1[n];
            #pragma unroll
            for (int tt = 0; tt < TILES; ++tt)
                #pragma unroll
                for (int r = 0; r < 8; ++r) {
                    float v = acc[tt][nt][r] + bias;
                    v = v > 0.0f ? v : 0.0f;
                    xw[(tt * 16 + r + 8 * hfid) * RS_X1 + n] = (_Float16)v;
                }
        }
    }
    __syncthreads();

    // ---- Layer 2: [16x128]@[128x64], B fragments shared across both tiles ----
    v8f acc2[TILES][4];
    #pragma unroll
    for (int tt = 0; tt < TILES; ++tt)
        #pragma unroll
        for (int nt = 0; nt < 4; ++nt) acc2[tt][nt] = (v8f){};
    {
        const _Float16* a2[TILES];
        #pragma unroll
        for (int tt = 0; tt < TILES; ++tt)
            a2[tt] = x1s + ((wave * TILES + tt) * 16 + sub) * RS_X1;
        #pragma unroll
        for (int kk = 0; kk < 4; ++kk) {
            const int ka = kk * 32 + hfid * 8;
            v16h a0 = comb16(*(const v8h*)(a2[0] + ka),
                             *(const v8h*)(a2[0] + ka + 16));
            v16h a1 = comb16(*(const v8h*)(a2[1] + ka),
                             *(const v8h*)(a2[1] + ka + 16));
            const int kb = kk * 32 + hfid * 8;
            #pragma unroll
            for (int nt = 0; nt < 4; ++nt) {
                const _Float16* brow = w2t + (nt * 16 + sub) * RS_W2;
                v16h b = comb16(*(const v8h*)(brow + kb),
                                *(const v8h*)(brow + kb + 16));
                acc2[0][nt] = __builtin_amdgcn_wmma_f32_16x16x32_f16(
                    false, a0, false, b, (short)0, acc2[0][nt], false, false);
                acc2[1][nt] = __builtin_amdgcn_wmma_f32_16x16x32_f16(
                    false, a1, false, b, (short)0, acc2[1][nt], false, false);
            }
        }
    }

    // ---- Layer 3: relu(x2) . W3 + b3; half-wave shuffle reduction ----
    const float b3v = sb3[0];
    #pragma unroll
    for (int tt = 0; tt < TILES; ++tt) {
        float tsum[8];
        #pragma unroll
        for (int r = 0; r < 8; ++r) {
            float s3 = 0.0f;
            #pragma unroll
            for (int nt = 0; nt < 4; ++nt) {
                const int n = nt * 16 + sub;
                float v = acc2[tt][nt][r] + sb2[n];
                v = v > 0.0f ? v : 0.0f;
                s3 += v * sw3[n];
            }
            tsum[r] = s3;
        }
        #pragma unroll
        for (int r = 0; r < 8; ++r) {
            #pragma unroll
            for (int mask = 1; mask < 16; mask <<= 1)
                tsum[r] += __shfl_xor(tsum[r], mask, 32);
        }
        if (sub == 0) {
            #pragma unroll
            for (int r = 0; r < 8; ++r) {
                const int pp = (wtile + tt) * 16 + r + 8 * hfid;
                if (pp < n_pairs) OUT[pp] = tsum[r] + b3v;
            }
        }
    }
}

extern "C" void kernel_launch(void* const* d_in, const int* in_sizes, int n_in,
                              void* d_out, int out_size, void* d_ws, size_t ws_size,
                              hipStream_t stream) {
    (void)n_in; (void)d_ws; (void)ws_size; (void)out_size;
    const float*     H  = (const float*)d_in[0];
    const long long* P  = (const long long*)d_in[1];
    const float*     W1 = (const float*)d_in[2];
    const float*     B1 = (const float*)d_in[3];
    const float*     W2 = (const float*)d_in[4];
    const float*     B2 = (const float*)d_in[5];
    const float*     W3 = (const float*)d_in[6];
    const float*     B3 = (const float*)d_in[7];
    float* OUT = (float*)d_out;

    const int n_pairs = in_sizes[1] / 2;
    if (n_pairs <= 0) return;

    // 1) weight prep (f32 -> f16 transposed image in device-global memory)
    prep_weights<<<dim3(64), dim3(256), 0, stream>>>(W1, W2);

    // 2) main fused gather + MLP kernel, ~319.7 KB dynamic LDS per workgroup
    (void)hipFuncSetAttribute((const void*)pair_mlp_wmma,
                              hipFuncAttributeMaxDynamicSharedMemorySize,
                              (int)LDS_BYTES);
    const int pairsPerBlock = WAVES * TILES * 16;  // 128
    dim3 grid((n_pairs + pairsPerBlock - 1) / pairsPerBlock);
    dim3 block(WAVES * 32);
    pair_mlp_wmma<<<grid, block, LDS_BYTES, stream>>>(H, P, B1, B2, W3, B3,
                                                      OUT, n_pairs);
}
